// Completion_37907381354607
// MI455X (gfx1250) — matrix operand order repaired and checked by reading, not compiled, not run on hardware
//
#include <hip/hip_runtime.h>
#include <hip/hip_bf16.h>

// Problem constants (from the reference)
#define Bsz 512
#define Tst 75
#define INd 96
#define Hd  1024
#define BH  (Bsz * Hd)

typedef __bf16 v16bf __attribute__((ext_vector_type(16)));
typedef float  v8f   __attribute__((ext_vector_type(8)));

union ABu { v16bf v; uint4 u[2]; };

// A-matrix (16x32 bf16) per-lane load, CDNA5 wave32 layout:
// lane<16: elems 0..7 = K[0..7],  8..15 = K[16..23]   (row M = lane)
// lane>=16: elems 0..7 = K[8..15], 8..15 = K[24..31]  (row M = lane-16)
__device__ __forceinline__ v16bf load_a(const __hip_bfloat16* row, int k0, int half) {
  ABu t;
  t.u[0] = *reinterpret_cast<const uint4*>(row + k0 + half * 8);
  t.u[1] = *reinterpret_cast<const uint4*>(row + k0 + 16 + half * 8);
  return t.v;
}
// B-matrix (32x16 bf16) per-lane load: lane holds column N = lane%16,
// elems 0..7 = K[half*8 + 0..7], 8..15 = K[16 + half*8 + 0..7] (same K order as A).
__device__ __forceinline__ v16bf load_bm(const __hip_bfloat16* row, int k0, int half) {
  ABu t;
  t.u[0] = *reinterpret_cast<const uint4*>(row + k0 + half * 8);
  t.u[1] = *reinterpret_cast<const uint4*>(row + k0 + 16 + half * 8);
  return t.v;
}

#define WMMA_BF16(A, Bm, C) \
  __builtin_amdgcn_wmma_f32_16x16x32_bf16(false, (A), false, (Bm), (short)0, (C), false, false)

__device__ __forceinline__ float sigm(float x)     { return 1.0f / (1.0f + __expf(-x)); }
__device__ __forceinline__ float tanhfast(float x) { return 2.0f / (1.0f + __expf(-2.0f * x)) - 1.0f; }

// ---------------------------------------------------------------------------
// Fused LSTM layer: gates = [X|Hin] @ [Wih|Whh]^T + bias, then cell update.
// Wave tile: 32 batch rows x 16 h-columns, all 4 gates (8 accumulators).
// Block tile: 128 batch rows x 32 h-cols (4 M-waves x 2 J-waves) to balance
// unique A/B bytes pulled through WGP$ (~0.5 MB each per block).
// Grid: 1024 waves = 128 blocks x 8 waves.
// ---------------------------------------------------------------------------
__global__ __launch_bounds__(256) void lstm_layer_kernel(
    const __hip_bfloat16* __restrict__ X, int Kx,       // (B, Kx) layer input, bf16
    const __hip_bfloat16* __restrict__ Wih,             // (4H, Kx) bf16
    const __hip_bfloat16* __restrict__ Hin,             // (B, H) bf16
    const __hip_bfloat16* __restrict__ Whh,             // (4H, H) bf16
    const float* __restrict__ bias,                     // (4H) = bih + bhh
    float* __restrict__ Cst,                            // (B, H) cell state, in/out
    __hip_bfloat16* __restrict__ Hout)                  // (B, H) new hidden, bf16
{
  const int lane = threadIdx.x & 31;
  const int wave = threadIdx.x >> 5;
  // 2-D block tiling: grid 128 blocks = 4 M-groups x 32 J-groups.
  const int mg = blockIdx.x >> 5;              // [0,4)   128-row block stripe
  const int jg = blockIdx.x & 31;              // [0,32)  32-col block stripe
  const int m0 = mg * 128 + (wave >> 1) * 32;  // wave's 32-row tile
  const int j0 = jg * 32 + (wave & 1) * 16;    // wave's 16-col tile
  const int half = lane >> 4;
  const int lr   = lane & 15;

  v8f c00 = {}, c01 = {}, c10 = {}, c11 = {};
  v8f c20 = {}, c21 = {}, c30 = {}, c31 = {};

  // ---- X @ Wih^T ----
  {
    const __hip_bfloat16* xr0 = X + (size_t)(m0 + lr) * Kx;
    const __hip_bfloat16* xr1 = X + (size_t)(m0 + 16 + lr) * Kx;
    const __hip_bfloat16* w0  = Wih + (size_t)(0 * Hd + j0 + lr) * Kx;
    const __hip_bfloat16* w1  = Wih + (size_t)(1 * Hd + j0 + lr) * Kx;
    const __hip_bfloat16* w2  = Wih + (size_t)(2 * Hd + j0 + lr) * Kx;
    const __hip_bfloat16* w3  = Wih + (size_t)(3 * Hd + j0 + lr) * Kx;
    for (int k0 = 0; k0 < Kx; k0 += 32) {
      v16bf a0 = load_a(xr0, k0, half);
      v16bf a1 = load_a(xr1, k0, half);
      v16bf b;
      b = load_bm(w0, k0, half); c00 = WMMA_BF16(a0, b, c00); c01 = WMMA_BF16(a1, b, c01);
      b = load_bm(w1, k0, half); c10 = WMMA_BF16(a0, b, c10); c11 = WMMA_BF16(a1, b, c11);
      b = load_bm(w2, k0, half); c20 = WMMA_BF16(a0, b, c20); c21 = WMMA_BF16(a1, b, c21);
      b = load_bm(w3, k0, half); c30 = WMMA_BF16(a0, b, c30); c31 = WMMA_BF16(a1, b, c31);
    }
  }
  // ---- Hin @ Whh^T ----
  {
    const __hip_bfloat16* xr0 = Hin + (size_t)(m0 + lr) * Hd;
    const __hip_bfloat16* xr1 = Hin + (size_t)(m0 + 16 + lr) * Hd;
    const __hip_bfloat16* w0  = Whh + (size_t)(0 * Hd + j0 + lr) * Hd;
    const __hip_bfloat16* w1  = Whh + (size_t)(1 * Hd + j0 + lr) * Hd;
    const __hip_bfloat16* w2  = Whh + (size_t)(2 * Hd + j0 + lr) * Hd;
    const __hip_bfloat16* w3  = Whh + (size_t)(3 * Hd + j0 + lr) * Hd;
    for (int k0 = 0; k0 < Hd; k0 += 32) {
      v16bf a0 = load_a(xr0, k0, half);
      v16bf a1 = load_a(xr1, k0, half);
      v16bf b;
      b = load_bm(w0, k0, half); c00 = WMMA_BF16(a0, b, c00); c01 = WMMA_BF16(a1, b, c01);
      b = load_bm(w1, k0, half); c10 = WMMA_BF16(a0, b, c10); c11 = WMMA_BF16(a1, b, c11);
      b = load_bm(w2, k0, half); c20 = WMMA_BF16(a0, b, c20); c21 = WMMA_BF16(a1, b, c21);
      b = load_bm(w3, k0, half); c30 = WMMA_BF16(a0, b, c30); c31 = WMMA_BF16(a1, b, c31);
    }
  }

  // ---- fused LSTM cell epilogue (C/D layout: VGPR v -> row half*8+v, col lane%16) ----
  const int j = j0 + lr;
  const float bi = bias[j];
  const float bf_ = bias[Hd + j];
  const float bg = bias[2 * Hd + j];
  const float bo = bias[3 * Hd + j];
#pragma unroll
  for (int v = 0; v < 8; ++v) {
    {
      const int m = m0 + half * 8 + v;
      const size_t idx = (size_t)m * Hd + j;
      const float cold = Cst[idx];
      const float iv = sigm(c00[v] + bi);
      const float fv = sigm(c10[v] + bf_);
      const float gv = tanhfast(c20[v] + bg);
      const float ov = sigm(c30[v] + bo);
      const float cn = fv * cold + iv * gv;
      Cst[idx] = cn;
      Hout[idx] = __float2bfloat16(ov * tanhfast(cn));
    }
    {
      const int m = m0 + 16 + half * 8 + v;
      const size_t idx = (size_t)m * Hd + j;
      const float cold = Cst[idx];
      const float iv = sigm(c01[v] + bi);
      const float fv = sigm(c11[v] + bf_);
      const float gv = tanhfast(c21[v] + bg);
      const float ov = sigm(c31[v] + bo);
      const float cn = fv * cold + iv * gv;
      Cst[idx] = cn;
      Hout[idx] = __float2bfloat16(ov * tanhfast(cn));
    }
  }
}

// ---------------------------------------------------------------------------
// FC head: out = clip(H2 @ fc_w^T + fc_b, -1, 1); writes fp32 output slice and
// bf16 feedback buffer for the next step's layer-0 input.
// Grid: 96 waves = 12 blocks x 8 waves (16 M-tiles x 6 N-tiles).
// ---------------------------------------------------------------------------
__global__ __launch_bounds__(256) void fc_kernel(
    const __hip_bfloat16* __restrict__ H2,    // (B, H) bf16
    const __hip_bfloat16* __restrict__ Wfc,   // (IN, H) bf16
    const float* __restrict__ bfc,            // (IN)
    float* __restrict__ outT,                 // d_out + t*IN ; row stride Tst*INd
    __hip_bfloat16* __restrict__ Xnext)       // (B, IN) bf16
{
  const int lane = threadIdx.x & 31;
  const int wave = threadIdx.x >> 5;
  const int tile = blockIdx.x * 8 + wave;     // [0, 96)
  const int m0   = (tile / 6) * 32;
  const int n0   = (tile % 6) * 16;
  const int half = lane >> 4;
  const int lr   = lane & 15;

  v8f c0 = {}, c1 = {};
  const __hip_bfloat16* xr0 = H2 + (size_t)(m0 + lr) * Hd;
  const __hip_bfloat16* xr1 = H2 + (size_t)(m0 + 16 + lr) * Hd;
  const __hip_bfloat16* wr  = Wfc + (size_t)(n0 + lr) * Hd;
  for (int k0 = 0; k0 < Hd; k0 += 32) {
    v16bf a0 = load_a(xr0, k0, half);
    v16bf a1 = load_a(xr1, k0, half);
    v16bf b  = load_bm(wr, k0, half);
    c0 = WMMA_BF16(a0, b, c0);
    c1 = WMMA_BF16(a1, b, c1);
  }

  const int n = n0 + lr;
  const float bb = bfc[n];
#pragma unroll
  for (int v = 0; v < 8; ++v) {
    {
      const int m = m0 + half * 8 + v;
      float val = fminf(fmaxf(c0[v] + bb, -1.0f), 1.0f);
      outT[(size_t)m * (Tst * INd) + n] = val;
      Xnext[(size_t)m * INd + n] = __float2bfloat16(val);
    }
    {
      const int m = m0 + 16 + half * 8 + v;
      float val = fminf(fmaxf(c1[v] + bb, -1.0f), 1.0f);
      outT[(size_t)m * (Tst * INd) + n] = val;
      Xnext[(size_t)m * INd + n] = __float2bfloat16(val);
    }
  }
}

// ---------------------------------------------------------------------------
// Init kernels (run every call; deterministic)
// ---------------------------------------------------------------------------
__global__ void cvt_kernel(const float* __restrict__ s, __hip_bfloat16* __restrict__ d, int n) {
  int i = blockIdx.x * blockDim.x + threadIdx.x;
  if (i < n) d[i] = __float2bfloat16(s[i]);
}

__global__ void bias_kernel(const float* __restrict__ a, const float* __restrict__ b,
                            float* __restrict__ o, int n) {
  int i = blockIdx.x * blockDim.x + threadIdx.x;
  if (i < n) o[i] = a[i] + b[i];
}

__global__ void state_init_kernel(const float* __restrict__ hiddens,   // (3,B,H)
                                  const float* __restrict__ cells,     // (3,B,H)
                                  const float* __restrict__ inputs,    // (B,T,IN)
                                  __hip_bfloat16* __restrict__ h0,
                                  __hip_bfloat16* __restrict__ h1,
                                  __hip_bfloat16* __restrict__ h2,
                                  float* __restrict__ c,               // (3,B,H)
                                  __hip_bfloat16* __restrict__ x)      // (B,IN)
{
  int i = blockIdx.x * blockDim.x + threadIdx.x;
  if (i < 3 * BH) {
    c[i] = cells[i];
    int l = i / BH, r = i - l * BH;
    __hip_bfloat16 v = __float2bfloat16(hiddens[i]);
    if (l == 0) h0[r] = v; else if (l == 1) h1[r] = v; else h2[r] = v;
  }
  if (i < Bsz * INd) {
    int b = i / INd, n = i - b * INd;
    x[i] = __float2bfloat16(inputs[(size_t)b * (Tst * INd) + n]);  // inputs[b, 0, :]
  }
}

// ---------------------------------------------------------------------------
extern "C" void kernel_launch(void* const* d_in, const int* in_sizes, int n_in,
                              void* d_out, int out_size, void* d_ws, size_t ws_size,
                              hipStream_t stream) {
  const float* inputs  = (const float*)d_in[0];
  const float* hiddens = (const float*)d_in[1];
  const float* cells   = (const float*)d_in[2];
  const float* Wih[3]  = {(const float*)d_in[3], (const float*)d_in[7],  (const float*)d_in[11]};
  const float* Whh[3]  = {(const float*)d_in[4], (const float*)d_in[8],  (const float*)d_in[12]};
  const float* bih[3]  = {(const float*)d_in[5], (const float*)d_in[9],  (const float*)d_in[13]};
  const float* bhh[3]  = {(const float*)d_in[6], (const float*)d_in[10], (const float*)d_in[14]};
  const float* fc_w    = (const float*)d_in[15];
  const float* fc_b    = (const float*)d_in[16];
  float* out = (float*)d_out;

  // ---- workspace layout (~53 MB) ----
  char* ws = (char*)d_ws;
  size_t off = 0;
  auto alloc = [&](size_t bytes) -> char* {
    char* p = ws + off;
    off = (off + bytes + 255) & ~(size_t)255;
    return p;
  };
  __hip_bfloat16* wih_b[3];
  __hip_bfloat16* whh_b[3];
  wih_b[0] = (__hip_bfloat16*)alloc((size_t)4 * Hd * INd * 2);
  whh_b[0] = (__hip_bfloat16*)alloc((size_t)4 * Hd * Hd * 2);
  for (int l = 1; l < 3; ++l) {
    wih_b[l] = (__hip_bfloat16*)alloc((size_t)4 * Hd * Hd * 2);
    whh_b[l] = (__hip_bfloat16*)alloc((size_t)4 * Hd * Hd * 2);
  }
  __hip_bfloat16* fcw_b = (__hip_bfloat16*)alloc((size_t)INd * Hd * 2);
  float* bias[3];
  for (int l = 0; l < 3; ++l) bias[l] = (float*)alloc((size_t)4 * Hd * 4);
  float* cst = (float*)alloc((size_t)3 * BH * 4);
  __hip_bfloat16* hbuf[3][2];
  for (int l = 0; l < 3; ++l)
    for (int p = 0; p < 2; ++p) hbuf[l][p] = (__hip_bfloat16*)alloc((size_t)BH * 2);
  __hip_bfloat16* xb = (__hip_bfloat16*)alloc((size_t)Bsz * INd * 2);

  // ---- per-call init: bf16 weights, combined biases, state ----
  auto cvt = [&](const float* s, __hip_bfloat16* d, int n) {
    cvt_kernel<<<(n + 255) / 256, 256, 0, stream>>>(s, d, n);
  };
  cvt(Wih[0], wih_b[0], 4 * Hd * INd);
  cvt(Whh[0], whh_b[0], 4 * Hd * Hd);
  cvt(Wih[1], wih_b[1], 4 * Hd * Hd);
  cvt(Whh[1], whh_b[1], 4 * Hd * Hd);
  cvt(Wih[2], wih_b[2], 4 * Hd * Hd);
  cvt(Whh[2], whh_b[2], 4 * Hd * Hd);
  cvt(fc_w, fcw_b, INd * Hd);
  for (int l = 0; l < 3; ++l)
    bias_kernel<<<(4 * Hd + 255) / 256, 256, 0, stream>>>(bih[l], bhh[l], bias[l], 4 * Hd);
  state_init_kernel<<<(3 * BH + 255) / 256, 256, 0, stream>>>(
      hiddens, cells, inputs, hbuf[0][0], hbuf[1][0], hbuf[2][0], cst, xb);

  // ---- 75 autoregressive steps, 4 fused kernels each ----
  int cur = 0;
  for (int t = 0; t < Tst; ++t) {
    const int nxt = cur ^ 1;
    lstm_layer_kernel<<<128, 256, 0, stream>>>(xb, INd, wih_b[0], hbuf[0][cur], whh_b[0],
                                               bias[0], cst + 0 * BH, hbuf[0][nxt]);
    lstm_layer_kernel<<<128, 256, 0, stream>>>(hbuf[0][nxt], Hd, wih_b[1], hbuf[1][cur], whh_b[1],
                                               bias[1], cst + 1 * BH, hbuf[1][nxt]);
    lstm_layer_kernel<<<128, 256, 0, stream>>>(hbuf[1][nxt], Hd, wih_b[2], hbuf[2][cur], whh_b[2],
                                               bias[2], cst + 2 * BH, hbuf[2][nxt]);
    fc_kernel<<<12, 256, 0, stream>>>(hbuf[2][nxt], fcw_b, fc_b, out + (size_t)t * INd, xb);
    cur = nxt;
  }
}
